// TransformerEncoderLayerSparse_23149873725779
// MI455X (gfx1250) — matrix operand order repaired and checked by reading, not compiled, not run on hardware
//
#include <hip/hip_runtime.h>
#include <math.h>

// ---------------- CDNA5 WMMA types ----------------
typedef __attribute__((ext_vector_type(16))) __bf16 v16bf;
typedef __attribute__((ext_vector_type(8)))  float  v8f;
typedef __attribute__((ext_vector_type(4)))  unsigned u32x4;
typedef __attribute__((ext_vector_type(8)))  unsigned u32x8;

#define EMBED 512
#define HEADS 8
#define DH    64
#define FFDIM 2048
#define SEQ   2048
#define BATCH 8
#define MROWS (BATCH*SEQ)

struct alignas(16) B128v { unsigned x, y, z, w; };
union FragU { v16bf v; unsigned u[8]; };

// A-operand fragment (16xK row-major tile; rowp = this lane's row, row = lane & 15).
// lanes 0-15 hold K 0..15, lanes 16-31 hold K 16..31 (matches frag_b).
__device__ inline v16bf frag_a(const __bf16* rowp, int lane) {
  FragU r;
  int kb = (lane >> 4) << 4;               // 0 or 16
#pragma unroll
  for (int j = 0; j < 8; ++j) {
    int k0 = kb + (j << 1);                   // {0,16} + 2*j
    r.u[j] = *(const unsigned*)(rowp + k0);
  }
  return r.v;
}

// B-operand fragment (K x 16). rowp = this lane's N-column, contraction-major
// (pairs contiguous). lanes 0-15: K=0..15, lanes 16-31: K=16..31.
__device__ inline v16bf frag_b(const __bf16* rowp, int lane) {
  FragU r;
  int kb = (lane >> 4) << 4;               // 0 or 16
#pragma unroll
  for (int j = 0; j < 8; ++j) r.u[j] = *(const unsigned*)(rowp + kb + 2 * j);
  return r.v;
}

// ---------------- gfx1250 async / TDM data movement ----------------
// LDS aperture flat addresses: LDS address is the low 32 bits -> truncate generic ptr.
__device__ inline unsigned lds_addr_of(const void* p) {
  return (unsigned)(size_t)p;
}

// Direct global -> LDS async copy, 16B per lane, tracked by ASYNCcnt.
__device__ inline void async_copy_b128(void* lds, const void* g) {
  unsigned l = lds_addr_of(lds);
  unsigned long long a = (unsigned long long)(size_t)g;
  asm volatile("global_load_async_to_lds_b128 %0, %1, off"
               :: "v"(l), "v"(a) : "memory");
}
#define WAIT_ASYNCCNT(n) asm volatile("s_wait_asynccnt %0" :: "n"(n) : "memory")

// One TDM descriptor: load a 2D tile (tile0 elems/row x tile1 rows, 2-byte
// elements, row stride = stride elems) from global into LDS, packed contiguously.
// D# group0/group1 per CDNA5 ISA. Tracked by TENSORcnt.
__device__ inline void tdm_load_tile_2d(unsigned lds_addr, const void* gptr,
                                        unsigned tile0, unsigned tile1,
                                        unsigned long long stride_elems) {
  unsigned long long ga = (unsigned long long)(size_t)gptr;
  const unsigned TD0 = 0x7fffffffu, TD1 = 0x7fffffffu;  // huge tensor dims: no OOB clip
  u32x4 g0;
  g0[0] = 1u;                                   // count=1, user descriptor
  g0[1] = lds_addr;                             // lds_addr [63:32]
  g0[2] = (unsigned)ga;                         // global_addr [95:64]
  g0[3] = (unsigned)(ga >> 32) | 0x80000000u;   // global_addr hi + type=2 [127:126]
  u32x8 g1;
  g1[0] = 1u << 16;                             // workgroup_mask=0, data_size=1 (2B)
  g1[1] = (TD0 & 0xffffu) << 16;                // tensor_dim0[15:0] @ bits 63:48
  g1[2] = ((TD0 >> 16) & 0xffffu) | ((TD1 & 0xffffu) << 16);   // dim0 hi | dim1 lo
  g1[3] = ((TD1 >> 16) & 0xffffu) | (tile0 << 16);             // dim1 hi | tile_dim0
  g1[4] = tile1;                                // tile_dim1 | tile_dim2=0
  g1[5] = (unsigned)stride_elems;               // tensor_dim0_stride[31:0]
  g1[6] = (unsigned)(stride_elems >> 32) & 0xffffu;  // stride hi | dim1_stride lo = 0
  g1[7] = 0u;
  asm volatile("tensor_load_to_lds %0, %1" :: "s"(g0), "s"(g1) : "memory");
}

// ---------------- generic bf16 WMMA GEMM ----------------
// C[M,N] = A[M,K] * Bt[N,K]^T + bias, epilogue variants.
// A tile via per-lane async-to-LDS b128; B tile via one TDM descriptor (wave 0).
// Double-buffered LDS so tile t+1 streams in while tile t feeds the WMMAs.
#define BM 128
#define BN 128
#define BK 32
enum { MODE_F32 = 0, MODE_BF16 = 1, MODE_BF16_RELU = 2, MODE_BF16_VT = 3 };

__global__ __launch_bounds__(256) void gemm_bf16_wmma(
    const __bf16* __restrict__ A, const __bf16* __restrict__ Bt,
    const float* __restrict__ bias, void* __restrict__ C,
    int M, int N, int K, int mode)
{
  __shared__ __bf16 sA[2][BM * BK];
  __shared__ __bf16 sB[2][BN * BK];
  int tid  = threadIdx.x;
  int lane = tid & 31, wave = tid >> 5;
  int wm = wave >> 2, wn = wave & 3;                 // 2 x 4 wave grid
  int gm0 = blockIdx.x * BM, gn0 = blockIdx.y * BN;

  v8f zero = {};
  v8f acc[4][2];
#pragma unroll
  for (int i = 0; i < 4; ++i)
#pragma unroll
    for (int j = 0; j < 2; ++j) acc[i][j] = zero;

  int e0 = tid << 3;                               // 2 b128 chunks per thread per A tile
  int e1 = (tid + 256) << 3;
  int r0 = e0 >> 5, c0 = e0 & 31;
  int r1 = e1 >> 5, c1 = e1 & 31;

  // ---- prologue: stage tile 0 into buffer 0 ----
  async_copy_b128(&sA[0][e0], A + (size_t)(gm0 + r0) * K + c0);
  async_copy_b128(&sA[0][e1], A + (size_t)(gm0 + r1) * K + c1);
  if (wave == 0)
    tdm_load_tile_2d(lds_addr_of(&sB[0][0]), Bt + (size_t)gn0 * K,
                     BK, BN, (unsigned long long)K);

  int nt = K / BK;
  for (int t = 0; t < nt; ++t) {
    int buf = t & 1;
    if (t + 1 < nt) {
      int kb = (t + 1) * BK;
      async_copy_b128(&sA[buf ^ 1][e0], A + (size_t)(gm0 + r0) * K + kb + c0);
      async_copy_b128(&sA[buf ^ 1][e1], A + (size_t)(gm0 + r1) * K + kb + c1);
      if (wave == 0)
        tdm_load_tile_2d(lds_addr_of(&sB[buf ^ 1][0]), Bt + (size_t)gn0 * K + kb,
                         BK, BN, (unsigned long long)K);
      WAIT_ASYNCCNT(2);                            // tile t complete, t+1 in flight
      if (wave == 0) __builtin_amdgcn_s_wait_tensorcnt((short)1);
    } else {
      WAIT_ASYNCCNT(0);
      if (wave == 0) __builtin_amdgcn_s_wait_tensorcnt((short)0);
    }
    __syncthreads();                               // buffer `buf` ready for all waves

    v16bf af[4], bfrag[2];
#pragma unroll
    for (int i = 0; i < 4; ++i)
      af[i] = frag_a(&sA[buf][((wm * 4 + i) * 16 + (lane & 15)) * BK], lane);
#pragma unroll
    for (int j = 0; j < 2; ++j)
      bfrag[j] = frag_b(&sB[buf][((wn * 2 + j) * 16 + (lane & 15)) * BK], lane);

#pragma unroll
    for (int i = 0; i < 4; ++i)
#pragma unroll
      for (int j = 0; j < 2; ++j)
        acc[i][j] = __builtin_amdgcn_wmma_f32_16x16x32_bf16(
            false, af[i], false, bfrag[j], (short)0, acc[i][j], false, false);
    __syncthreads();                               // done reading `buf` (reused at t+2)
  }

  int hl = lane >> 4, nin = lane & 15;
#pragma unroll
  for (int i = 0; i < 4; ++i) {
#pragma unroll
    for (int j = 0; j < 2; ++j) {
      int gn = gn0 + (wn * 2 + j) * 16 + nin;
      float bb = bias ? bias[gn] : 0.0f;
#pragma unroll
      for (int r = 0; r < 8; ++r) {
        int gm = gm0 + (wm * 4 + i) * 16 + r + hl * 8;   // C-layout: VGPR r + lane-half
        float val = acc[i][j][r] + bb;
        if (mode == MODE_F32) {
          ((float*)C)[(size_t)gm * N + gn] = val;
        } else if (mode == MODE_BF16) {
          ((__bf16*)C)[(size_t)gm * N + gn] = (__bf16)val;
        } else if (mode == MODE_BF16_RELU) {
          ((__bf16*)C)[(size_t)gm * N + gn] = (__bf16)(val > 0.0f ? val : 0.0f);
        } else {                                         // V transposed per head: [B,H,dh,S]
          int b = gm >> 11, s = gm & 2047;
          int h = gn >> 6,  d = gn & 63;
          ((__bf16*)C)[((((size_t)b * HEADS + h) * DH + d) << 11) + s] = (__bf16)val;
        }
      }
    }
  }
}

// ---------------- banded attention, one wave per (b,h,16-query tile) ----------------
__global__ __launch_bounds__(32) void attn_wmma(
    const __bf16* __restrict__ Q, const __bf16* __restrict__ Km,
    const __bf16* __restrict__ Vt, __bf16* __restrict__ O)
{
  int qt = blockIdx.x, h = blockIdx.y, b = blockIdx.z;
  int lane = threadIdx.x;
  int q0 = qt << 4;
  int hl = lane >> 4, nin = lane & 15;

  // Q tile as A-operand: 16 x 64 = two K=32 fragments, straight from global.
  const __bf16* qrow = Q + ((size_t)(b * SEQ + q0 + nin) * EMBED) + h * DH;
  v16bf aq0 = frag_a(qrow, lane);
  v16bf aq1 = frag_a(qrow + 32, lane);

  int t0 = qt - 4;                       // band covers key tiles qt-4 .. qt+4
  v8f sc[9];

#pragma unroll
  for (int i = 0; i < 9; ++i) {
    int tt = t0 + i;
    if (tt >= 0 && tt < (SEQ >> 4)) {
      const __bf16* krow = Km + ((size_t)(b * SEQ + (tt << 4) + nin) * EMBED) + h * DH;
      v16bf bk0 = frag_b(krow, lane);        // contraction over dh, contiguous pairs
      v16bf bk1 = frag_b(krow + 32, lane);
      v8f c = {};
      c = __builtin_amdgcn_wmma_f32_16x16x32_bf16(false, aq0, false, bk0, (short)0, c, false, false);
      c = __builtin_amdgcn_wmma_f32_16x16x32_bf16(false, aq1, false, bk1, (short)0, c, false, false);
#pragma unroll
      for (int r = 0; r < 8; ++r) {
        float qi = (float)(q0 + r + hl * 8);
        float ki = (float)((tt << 4) + nin);
        float d  = qi - ki;
        // score/8 + gauss(sigma=2) + cos(2*pi*d/100), then window mask |d|<=64
        float s = c[r] * 0.125f + __expf(-d * d * 0.125f)
                + __cosf(0.06283185307179586f * d);
        if (fabsf(d) > 64.0f) s = -1e9f;
        sc[i][r] = s;
      }
    } else {
#pragma unroll
      for (int r = 0; r < 8; ++r) sc[i][r] = -1e9f;
    }
  }

  // row-wise softmax: row r+8*hl is spread over 16 lanes of a half -> shfl_xor 1..8
  float inv[8];
#pragma unroll
  for (int r = 0; r < 8; ++r) {
    float m = -1e30f;
#pragma unroll
    for (int i = 0; i < 9; ++i) m = fmaxf(m, sc[i][r]);
    m = fmaxf(m, __shfl_xor(m, 1, 32));
    m = fmaxf(m, __shfl_xor(m, 2, 32));
    m = fmaxf(m, __shfl_xor(m, 4, 32));
    m = fmaxf(m, __shfl_xor(m, 8, 32));
    float sum = 0.0f;
#pragma unroll
    for (int i = 0; i < 9; ++i) { float p = __expf(sc[i][r] - m); sc[i][r] = p; sum += p; }
    sum += __shfl_xor(sum, 1, 32);
    sum += __shfl_xor(sum, 2, 32);
    sum += __shfl_xor(sum, 4, 32);
    sum += __shfl_xor(sum, 8, 32);
    inv[r] = 1.0f / sum;
  }

  // P * V over key-tile pairs; P staged via LDS to re-layout C->A fragment.
  __shared__ __bf16 pls[16 * 32];
  v8f o[4];
  v8f z2 = {};
#pragma unroll
  for (int nn = 0; nn < 4; ++nn) o[nn] = z2;

  for (int pr = 0; pr < 5; ++pr) {
    int i0 = pr * 2, i1 = pr * 2 + 1;
#pragma unroll
    for (int r = 0; r < 8; ++r) {
      int m = r + hl * 8;
      pls[m * 32 + nin]      = (__bf16)(sc[i0][r] * inv[r]);
      pls[m * 32 + 16 + nin] = (i1 < 9) ? (__bf16)(sc[i1][r] * inv[r]) : (__bf16)0.0f;
    }
    __syncthreads();
    v16bf ap = frag_a(pls + nin * 32, lane);
    int kb0 = (t0 + i0) << 4;
#pragma unroll
    for (int nn = 0; nn < 4; ++nn) {
      const __bf16* vrow = Vt + ((((size_t)b * HEADS + h) * DH) + nn * 16 + nin) * SEQ;
      FragU rb;
      int cb = hl << 4;
#pragma unroll
      for (int j = 0; j < 8; ++j) {
        int key = kb0 + cb + 2 * j;
        rb.u[j] = (key >= 0 && key < SEQ) ? *(const unsigned*)(vrow + key) : 0u;
      }
      o[nn] = __builtin_amdgcn_wmma_f32_16x16x32_bf16(
          false, ap, false, rb.v, (short)0, o[nn], false, false);
    }
    __syncthreads();
  }

#pragma unroll
  for (int nn = 0; nn < 4; ++nn)
#pragma unroll
    for (int r = 0; r < 8; ++r) {
      int m = q0 + r + hl * 8;
      O[((size_t)(b * SEQ + m)) * EMBED + h * DH + nn * 16 + nin] = (__bf16)o[nn][r];
    }
}

// ---------------- elementwise / LN kernels ----------------
__global__ void cvt_f32_bf16(const float* __restrict__ x, __bf16* __restrict__ y, int n) {
  int i = blockIdx.x * 256 + threadIdx.x;
  if (i < n) y[i] = (__bf16)x[i];
}

__global__ void transpose_f32_bf16(const float* __restrict__ W, __bf16* __restrict__ Wt,
                                   int K, int N) {
  int i = blockIdx.x * 256 + threadIdx.x;
  if (i < K * N) {
    int n = i / K, k = i % K;
    Wt[i] = (__bf16)W[(size_t)k * N + n];     // Wt[n][k]
  }
}

__global__ __launch_bounds__(128) void ln1_gate_kernel(
    const float* __restrict__ src, const float* __restrict__ ap,
    const float* __restrict__ Wg, const float* __restrict__ bg,
    const float* __restrict__ g, const float* __restrict__ beta,
    float* __restrict__ x1f, __bf16* __restrict__ x1h)
{
  int row = blockIdx.x, tid = threadIdx.x;
  const float* xr = src + (size_t)row * EMBED;
  const float* ar = ap  + (size_t)row * EMBED;
  __shared__ float r1[128], r2[128];
  float xs[4], as2[4], gp = 0.0f;
#pragma unroll
  for (int i = 0; i < 4; ++i) {
    int c = i * 128 + tid;
    xs[i] = xr[c]; as2[i] = ar[c];
    gp += xs[i] * Wg[c];
  }
  r1[tid] = gp; __syncthreads();
  for (int s = 64; s > 0; s >>= 1) { if (tid < s) r1[tid] += r1[tid + s]; __syncthreads(); }
  float gate = 1.0f / (1.0f + __expf(-(r1[0] + bg[0])));
  gate = gate > 0.0f ? gate : 0.0f;            // THRESH = 0
  __syncthreads();
  float y[4], s1 = 0.0f, s2 = 0.0f;
#pragma unroll
  for (int i = 0; i < 4; ++i) { y[i] = xs[i] + gate * as2[i]; s1 += y[i]; s2 += y[i] * y[i]; }
  r1[tid] = s1; r2[tid] = s2; __syncthreads();
  for (int s = 64; s > 0; s >>= 1) {
    if (tid < s) { r1[tid] += r1[tid + s]; r2[tid] += r2[tid + s]; }
    __syncthreads();
  }
  float mean = r1[0] * (1.0f / EMBED);
  float var  = r2[0] * (1.0f / EMBED) - mean * mean;
  float rinv = rsqrtf(var + 1e-5f);
#pragma unroll
  for (int i = 0; i < 4; ++i) {
    int c = i * 128 + tid;
    float v = (y[i] - mean) * rinv * g[c] + beta[c];
    x1f[(size_t)row * EMBED + c] = v;
    x1h[(size_t)row * EMBED + c] = (__bf16)v;
  }
}

__global__ __launch_bounds__(128) void ln2_kernel(
    const float* __restrict__ x1, const float* __restrict__ ff2,
    const float* __restrict__ g, const float* __restrict__ beta,
    float* __restrict__ out)
{
  int row = blockIdx.x, tid = threadIdx.x;
  const float* xr = x1  + (size_t)row * EMBED;
  const float* fr = ff2 + (size_t)row * EMBED;
  __shared__ float r1[128], r2[128];
  float y[4], s1 = 0.0f, s2 = 0.0f;
#pragma unroll
  for (int i = 0; i < 4; ++i) {
    int c = i * 128 + tid;
    y[i] = xr[c] + fr[c]; s1 += y[i]; s2 += y[i] * y[i];
  }
  r1[tid] = s1; r2[tid] = s2; __syncthreads();
  for (int s = 64; s > 0; s >>= 1) {
    if (tid < s) { r1[tid] += r1[tid + s]; r2[tid] += r2[tid + s]; }
    __syncthreads();
  }
  float mean = r1[0] * (1.0f / EMBED);
  float var  = r2[0] * (1.0f / EMBED) - mean * mean;
  float rinv = rsqrtf(var + 1e-5f);
#pragma unroll
  for (int i = 0; i < 4; ++i) {
    int c = i * 128 + tid;
    out[(size_t)row * EMBED + c] = (y[i] - mean) * rinv * g[c] + beta[c];
  }
}

// ---------------- host orchestration ----------------
extern "C" void kernel_launch(void* const* d_in, const int* in_sizes, int n_in,
                              void* d_out, int out_size, void* d_ws, size_t ws_size,
                              hipStream_t stream) {
  (void)in_sizes; (void)n_in; (void)out_size; (void)ws_size;
  const float* src  = (const float*)d_in[0];
  const float* Wq   = (const float*)d_in[1];  const float* bq = (const float*)d_in[2];
  const float* Wk   = (const float*)d_in[3];  const float* bk = (const float*)d_in[4];
  const float* Wv   = (const float*)d_in[5];  const float* bv = (const float*)d_in[6];
  const float* Wo   = (const float*)d_in[7];  const float* bo = (const float*)d_in[8];
  const float* Wg   = (const float*)d_in[9];  const float* bg = (const float*)d_in[10];
  const float* W1   = (const float*)d_in[11]; const float* b1 = (const float*)d_in[12];
  const float* W2   = (const float*)d_in[13]; const float* b2 = (const float*)d_in[14];
  const float* ln1g = (const float*)d_in[15]; const float* ln1b = (const float*)d_in[16];
  const float* ln2g = (const float*)d_in[17]; const float* ln2b = (const float*)d_in[18];

  char* ws = (char*)d_ws;
  size_t off = 0;
  auto take = [&](size_t bytes) -> char* {
    char* p = ws + off;
    off += (bytes + 255) & ~(size_t)255;
    return p;
  };
  __bf16* Xb   = (__bf16*)take((size_t)MROWS * EMBED * 2);
  __bf16* WqT  = (__bf16*)take((size_t)EMBED * EMBED * 2);
  __bf16* WkT  = (__bf16*)take((size_t)EMBED * EMBED * 2);
  __bf16* WvT  = (__bf16*)take((size_t)EMBED * EMBED * 2);
  __bf16* WoT  = (__bf16*)take((size_t)EMBED * EMBED * 2);
  __bf16* W1T  = (__bf16*)take((size_t)EMBED * FFDIM * 2);
  __bf16* W2T  = (__bf16*)take((size_t)FFDIM * EMBED * 2);
  __bf16* Qb   = (__bf16*)take((size_t)MROWS * EMBED * 2);   // |
  __bf16* Kb   = (__bf16*)take((size_t)MROWS * EMBED * 2);   // | these 4 x 16MB are
  __bf16* Vtb  = (__bf16*)take((size_t)MROWS * EMBED * 2);   // | reused as ff1 (64MB)
  __bf16* Oh   = (__bf16*)take((size_t)MROWS * EMBED * 2);   // |
  float*  apro = (float*) take((size_t)MROWS * EMBED * 4);   // attn proj; reused as ff2
  float*  x1f  = (float*) take((size_t)MROWS * EMBED * 4);
  __bf16* x1h  = (__bf16*)take((size_t)MROWS * EMBED * 2);
  __bf16* ff1  = Qb;      // alias: Q/K/Vt/Oh dead after attention + Wo GEMM
  float*  ff2  = apro;    // alias: attn proj consumed by ln1 before FF2 runs

  int n = MROWS * EMBED;
  cvt_f32_bf16<<<n / 256, 256, 0, stream>>>(src, Xb, n);
  transpose_f32_bf16<<<(EMBED * EMBED) / 256, 256, 0, stream>>>(Wq, WqT, EMBED, EMBED);
  transpose_f32_bf16<<<(EMBED * EMBED) / 256, 256, 0, stream>>>(Wk, WkT, EMBED, EMBED);
  transpose_f32_bf16<<<(EMBED * EMBED) / 256, 256, 0, stream>>>(Wv, WvT, EMBED, EMBED);
  transpose_f32_bf16<<<(EMBED * EMBED) / 256, 256, 0, stream>>>(Wo, WoT, EMBED, EMBED);
  transpose_f32_bf16<<<(EMBED * FFDIM) / 256, 256, 0, stream>>>(W1, W1T, EMBED, FFDIM);
  transpose_f32_bf16<<<(FFDIM * EMBED) / 256, 256, 0, stream>>>(W2, W2T, FFDIM, EMBED);

  dim3 g512(MROWS / BM, EMBED / BN);
  gemm_bf16_wmma<<<g512, 256, 0, stream>>>(Xb, WqT, bq, Qb,  MROWS, EMBED, EMBED, MODE_BF16);
  gemm_bf16_wmma<<<g512, 256, 0, stream>>>(Xb, WkT, bk, Kb,  MROWS, EMBED, EMBED, MODE_BF16);
  gemm_bf16_wmma<<<g512, 256, 0, stream>>>(Xb, WvT, bv, Vtb, MROWS, EMBED, EMBED, MODE_BF16_VT);

  attn_wmma<<<dim3(SEQ / 16, HEADS, BATCH), 32, 0, stream>>>(Qb, Kb, Vtb, Oh);

  gemm_bf16_wmma<<<g512, 256, 0, stream>>>(Oh, WoT, bo, apro, MROWS, EMBED, EMBED, MODE_F32);
  ln1_gate_kernel<<<MROWS, 128, 0, stream>>>(src, apro, Wg, bg, ln1g, ln1b, x1f, x1h);

  gemm_bf16_wmma<<<dim3(MROWS / BM, FFDIM / BN), 256, 0, stream>>>(
      x1h, W1T, b1, ff1, MROWS, FFDIM, EMBED, MODE_BF16_RELU);
  gemm_bf16_wmma<<<g512, 256, 0, stream>>>(ff1, W2T, b2, ff2, MROWS, EMBED, FFDIM, MODE_F32);

  ln2_kernel<<<MROWS, 128, 0, stream>>>(x1f, ff2, ln2g, ln2b, (float*)d_out);
}
